// inforGCN_58394375357109
// MI455X (gfx1250) — matrix operand order repaired and checked by reading, not compiled, not run on hardware
//
#include <hip/hip_runtime.h>
#include <math.h>

typedef __attribute__((ext_vector_type(16))) _Float16 v16h;
typedef __attribute__((ext_vector_type(8)))  float    v8f;

static constexpr int    NU_   = 100000;
static constexpr int    NI_   = 50000;
static constexpr int    D_    = 64;
static constexpr int    NN_   = 150000;     // NU+NI
static constexpr int    NNZ_  = 2400000;
static constexpr int    NB_   = 8192;       // rating batch
static constexpr int    NBRI_ = 4096;
static constexpr int    BRI0_ = 50000;

__device__ __forceinline__ float logsig(float x) {
  // log(sigmoid(x)) = min(x,0) - log1p(exp(-|x|)); log(1-sigmoid(x)) = logsig(-x)
  return fminf(x, 0.f) - log1pf(__expf(-fabsf(x)));
}
__device__ __forceinline__ float sigm(float x) { return 1.f / (1.f + __expf(-x)); }

// ---------------- GCN propagation kernels ----------------

__global__ void k_init(const float* __restrict__ ue, const float* __restrict__ ie,
                       float* __restrict__ h, float* __restrict__ acc) {
  size_t g4 = (size_t)blockIdx.x * blockDim.x + threadIdx.x;   // over N*D/4
  size_t base = g4 * 4;
  float4 v = (base < (size_t)NU_ * D_)
                 ? ((const float4*)ue)[g4]
                 : ((const float4*)ie)[g4 - (size_t)NU_ * D_ / 4];
  ((float4*)h)[g4] = v;
  ((float4*)acc)[g4] = v;
}

// COO SpMM: y[r] += v * x[c]; one thread per (edge, 4-dim chunk); x/y are L2-resident (38MB << 192MB L2)
__global__ void k_spmm(const int* __restrict__ rows, const int* __restrict__ cols,
                       const float* __restrict__ vals, const float* __restrict__ x,
                       float* __restrict__ y) {
  int gid = blockIdx.x * 256 + threadIdx.x;   // NNZ*16 threads
  int e = gid >> 4, ch = gid & 15;
  int r = rows[e], c = cols[e];
  float v = vals[e];
  float4 xv = ((const float4*)x)[(size_t)c * 16 + ch];
  float* yp = y + (size_t)r * 64 + ch * 4;
  atomicAdd(yp + 0, v * xv.x);
  atomicAdd(yp + 1, v * xv.y);
  atomicAdd(yp + 2, v * xv.z);
  atomicAdd(yp + 3, v * xv.w);
}

// h = t + h ; acc += h
__global__ void k_add_acc(float* __restrict__ h, const float* __restrict__ t,
                          float* __restrict__ acc) {
  size_t g4 = (size_t)blockIdx.x * blockDim.x + threadIdx.x;
  float4 hv = ((float4*)h)[g4];
  float4 tv = ((const float4*)t)[g4];
  float4 av = ((float4*)acc)[g4];
  hv.x += tv.x; hv.y += tv.y; hv.z += tv.z; hv.w += tv.w;
  av.x += hv.x; av.y += hv.y; av.z += hv.z; av.w += hv.w;
  ((float4*)h)[g4] = hv;
  ((float4*)acc)[g4] = av;
}

__global__ void k_scale(float* __restrict__ acc) {
  size_t g4 = (size_t)blockIdx.x * blockDim.x + threadIdx.x;
  float4 v = ((float4*)acc)[g4];
  v.x *= 0.25f; v.y *= 0.25f; v.z *= 0.25f; v.w *= 0.25f;
  ((float4*)acc)[g4] = v;
}

// ---------------- BPR rating loss via WMMA (diag of U * P^T) ----------------
// One wave = 16 samples. A = 16 user rows (16x64), B = 16 item rows transposed (64x16),
// two v_wmma_f32_16x16x32_f16 accumulate; diagonal of 16x16 result = the 16 dot products.
__global__ void k_rating_wmma(const float* __restrict__ F1,
                              const int* __restrict__ user, const int* __restrict__ pos,
                              const int* __restrict__ neg, float* __restrict__ S0) {
  int wave = blockIdx.x * (blockDim.x >> 5) + (threadIdx.x >> 5);
  int lane = threadIdx.x & 31;
  int m = lane & 15;
  int s = wave * 16 + m;

  const float* uP = F1 + (size_t)user[s] * 64;
  const float* pP = F1 + ((size_t)NU_ + pos[s]) * 64;
  const float* nP = F1 + ((size_t)NU_ + neg[s]) * 64;

  // A layout (16-bit 16x32): lanes 0-15 hold K in {0..7,16..23}; lanes 16-31 hold {8..15,24..31}
  int ka = (lane < 16) ? 0 : 8;
  // B layout (16-bit 32x16): lanes 0-15 hold K in {0..7,16..23}; lanes 16-31 hold {8..15,24..31}; col = lane&15
  int kb = (lane < 16) ? 0 : 8;

  v16h a0, a1, bp0, bp1, bn0, bn1;
  float sq = 0.f;   // partial sum of squares for the L2 term (f32 source values)
#pragma unroll
  for (int i = 0; i < 8; ++i) {
    float u0 = uP[ka + i], u1 = uP[ka + 16 + i];
    float u2 = uP[ka + 32 + i], u3 = uP[ka + 48 + i];
    a0[i] = (_Float16)u0;  a0[8 + i] = (_Float16)u1;
    a1[i] = (_Float16)u2;  a1[8 + i] = (_Float16)u3;
    sq += u0 * u0 + u1 * u1 + u2 * u2 + u3 * u3;
  }
#pragma unroll
  for (int i = 0; i < 16; ++i) {
    int kk = kb + i + ((i >= 8) ? 8 : 0);
    float p0 = pP[kk], p1 = pP[kk + 32];
    float q0 = nP[kk], q1 = nP[kk + 32];
    bp0[i] = (_Float16)p0;  bp1[i] = (_Float16)p1;
    bn0[i] = (_Float16)q0;  bn1[i] = (_Float16)q1;
    sq += p0 * p0 + p1 * p1 + q0 * q0 + q1 * q1;
  }

  v8f cp = {}; v8f cn = {};
  cp = __builtin_amdgcn_wmma_f32_16x16x32_f16(false, a0, false, bp0, (short)0, cp, false, false);
  cp = __builtin_amdgcn_wmma_f32_16x16x32_f16(false, a1, false, bp1, (short)0, cp, false, false);
  cn = __builtin_amdgcn_wmma_f32_16x16x32_f16(false, a0, false, bn0, (short)0, cn, false, false);
  cn = __builtin_amdgcn_wmma_f32_16x16x32_f16(false, a1, false, bn1, (short)0, cn, false, false);

  // lanes l and l^16 jointly hold all 64 dims of sample m: combine squares across the pair
  float sqFull = sq + __shfl_xor(sq, 16, 32);

  float contrib = 0.f;
  if (lane < 16) contrib += 0.01f * sqFull;       // count L2 once per sample

  // Diagonal holders: lanes 0-7 -> c[lane] (M=N=lane); lanes 24-31 -> c[lane-24] (M=N=lane-16)
  int j = (lane < 8) ? lane : ((lane >= 24) ? (lane - 24) : -1);
  if (j >= 0) {
    float dp = 0.f, dn = 0.f;
#pragma unroll
    for (int i = 0; i < 8; ++i) { if (i == j) { dp = cp[i]; dn = cn[i]; } }
    contrib += -logsig(dp - dn);                  // -log_sigmoid(pos - neg)
  }
  atomicAdd(S0, contrib);
}

// ---------------- DGI loss reductions ----------------

// sum_{b,d} logsig(eReal[idx[b],d]) and sum_{b,d} logsig(-eFake[idx[b],d])
__global__ void k_dgi_pair(const float* __restrict__ eReal, const float* __restrict__ eFake,
                           const int* __restrict__ idx, float* sReal, float* sFake) {
  __shared__ float red[2][256];
  int t = threadIdx.x;
  int g = blockIdx.x * 256 + t;       // NBRI*64 threads
  int b = g >> 6, d = g & 63;
  size_t row = (size_t)idx[b] * 64 + d;
  red[0][t] = logsig(eReal[row]);
  red[1][t] = logsig(-eFake[row]);
  __syncthreads();
  for (int k = 128; k > 0; k >>= 1) {
    if (t < k) { red[0][t] += red[0][t + k]; red[1][t] += red[1][t + k]; }
    __syncthreads();
  }
  if (t == 0) { atomicAdd(sReal, red[0][0]); atomicAdd(sFake, red[1][0]); }
}

// column mean over contiguous rows -> outMean[d] (g_soc)
__global__ void k_colmean(const float* __restrict__ emb, int rowStart, int nRows,
                          float* __restrict__ outMean) {
  __shared__ float red[256];
  int d = blockIdx.x, t = threadIdx.x;
  float s = 0.f;
  for (int i = t; i < nRows; i += 256) s += emb[(size_t)(rowStart + i) * 64 + d];
  red[t] = s; __syncthreads();
  for (int k = 128; k > 0; k >>= 1) { if (t < k) red[t] += red[t + k]; __syncthreads(); }
  if (t == 0) outMean[d] = red[0] / (float)nRows;
}

// g_iu[d] = mean_b sigmoid(F4[i_bri[b],d]); g_ii[d] = mean_b sigmoid(F4[NU+i_bri_pos[b],d])
__global__ void k_gmeans_sig(const float* __restrict__ f4, const int* __restrict__ ibri,
                             const int* __restrict__ ibripos, float* __restrict__ giu,
                             float* __restrict__ gii) {
  __shared__ float red[2][256];
  int d = blockIdx.x, t = threadIdx.x;
  float s1 = 0.f, s2 = 0.f;
  for (int i = t; i < NBRI_; i += 256) {
    s1 += sigm(f4[(size_t)ibri[i] * 64 + d]);
    s2 += sigm(f4[((size_t)NU_ + ibripos[i]) * 64 + d]);
  }
  red[0][t] = s1; red[1][t] = s2; __syncthreads();
  for (int k = 128; k > 0; k >>= 1) {
    if (t < k) { red[0][t] += red[0][t + k]; red[1][t] += red[1][t + k]; }
    __syncthreads();
  }
  if (t == 0) { giu[d] = red[0][0] / (float)NBRI_; gii[d] = red[1][0] / (float)NBRI_; }
}

// S3 = sum_d log sigmoid(g_soc[d]); S4 = sum_d log(1 - sigmoid(g_soc[d]))
__global__ void k_gterms_soc(const float* __restrict__ g, float* s3, float* s4) {
  __shared__ float red[2][64];
  int t = threadIdx.x;
  float x = g[t];
  red[0][t] = logsig(x); red[1][t] = logsig(-x); __syncthreads();
  for (int k = 32; k > 0; k >>= 1) {
    if (t < k) { red[0][t] += red[0][t + k]; red[1][t] += red[1][t + k]; }
    __syncthreads();
  }
  if (t == 0) { *s3 = red[0][0]; *s4 = red[1][0]; }
}

// S7 = sum_d log(g_iu)+log(g_ii); S8 = sum_d log(1-g_iu)+log(1-g_ii)  (values already in (0,1))
__global__ void k_gterms_infor(const float* __restrict__ giu, const float* __restrict__ gii,
                               float* s7, float* s8) {
  __shared__ float red[2][64];
  int t = threadIdx.x;
  float a = giu[t], b = gii[t];
  red[0][t] = logf(a) + logf(b);
  red[1][t] = logf(1.f - a) + logf(1.f - b);
  __syncthreads();
  for (int k = 32; k > 0; k >>= 1) {
    if (t < k) { red[0][t] += red[0][t + k]; red[1][t] += red[1][t + k]; }
    __syncthreads();
  }
  if (t == 0) { *s7 = red[0][0]; *s8 = red[1][0]; }
}

__global__ void k_finalize(const float* __restrict__ s, float* __restrict__ out) {
  float rating = s[0];
  // social: real/fake each averaged over 4096 x 128 elements; summary half contributes 4096*Sg
  float social = -((s[1] + 4096.f * s[3]) + (s[2] + 4096.f * s[4])) / 524288.f;
  // infor: averaged over 4096 x 192 elements
  float infor  = -((s[5] + 4096.f * s[7]) + (s[6] + 4096.f * s[8])) / 786432.f;
  out[0] = rating + 100.f * social + 1000.f * infor;
  out[1] = rating;
  out[2] = social;
  out[3] = infor;
}

// ---------------- launch ----------------

extern "C" void kernel_launch(void* const* d_in, const int* in_sizes, int n_in,
                              void* d_out, int out_size, void* d_ws, size_t ws_size,
                              hipStream_t stream) {
  // setup_inputs() dict order:
  // 0 ue, 1 ie, 2 fue, 3 fie, 4 sgu, 5 sgi, 6 igu, 7 igi,
  // 8 r1, 9 c1, 10 v1, 11 r2, 12 c2, 13 v2, 14 r3, 15 c3, 16 v3, 17 r4, 18 c4, 19 v4,
  // 20 user, 21 pos, 22 neg, 23 s_bri, 24 i_bri, 25 i_bri_pos, 26 i_bri_neg (unused)
  const float* embU[4] = {(const float*)d_in[0], (const float*)d_in[2],
                          (const float*)d_in[4], (const float*)d_in[6]};
  const float* embI[4] = {(const float*)d_in[1], (const float*)d_in[3],
                          (const float*)d_in[5], (const float*)d_in[7]};
  const int*   rA[4] = {(const int*)d_in[8],  (const int*)d_in[11],
                        (const int*)d_in[14], (const int*)d_in[17]};
  const int*   cA[4] = {(const int*)d_in[9],  (const int*)d_in[12],
                        (const int*)d_in[15], (const int*)d_in[18]};
  const float* vA[4] = {(const float*)d_in[10], (const float*)d_in[13],
                        (const float*)d_in[16], (const float*)d_in[19]};
  const int* user  = (const int*)d_in[20];
  const int* pos   = (const int*)d_in[21];
  const int* neg   = (const int*)d_in[22];
  const int* s_bri = (const int*)d_in[23];
  const int* i_bri = (const int*)d_in[24];
  const int* i_bri_pos = (const int*)d_in[25];

  const size_t ND = (size_t)NN_ * D_;          // 9.6M floats
  float* bufH = (float*)d_ws;
  float* bufT = bufH + ND;
  float* F[4];
  for (int i = 0; i < 4; ++i) F[i] = bufH + (size_t)(2 + i) * ND;
  float* scal = bufH + 6 * ND;                 // [0..8]=sums, [16..79]=gsoc, [80..143]=giu, [144..207]=gii

  hipMemsetAsync(scal, 0, 256 * sizeof(float), stream);

  const int nd4Blocks   = (int)(ND / 4 / 256);       // 9375
  const int spmmBlocks  = NNZ_ * 16 / 256;           // 150000

  for (int L = 0; L < 4; ++L) {
    k_init<<<nd4Blocks, 256, 0, stream>>>(embU[L], embI[L], bufH, F[L]);
    for (int hop = 0; hop < 3; ++hop) {
      hipMemsetAsync(bufT, 0, ND * sizeof(float), stream);
      k_spmm<<<spmmBlocks, 256, 0, stream>>>(rA[L], cA[L], vA[L], bufH, bufT);
      k_add_acc<<<nd4Blocks, 256, 0, stream>>>(bufH, bufT, F[L]);
    }
    k_scale<<<nd4Blocks, 256, 0, stream>>>(F[L]);
  }

  // losses
  k_rating_wmma<<<NB_ / 16 / 8, 256, 0, stream>>>(F[0], user, pos, neg, &scal[0]);  // 64 blocks
  k_dgi_pair<<<NBRI_ * 64 / 256, 256, 0, stream>>>(F[0], F[1], s_bri, &scal[1], &scal[2]);
  k_colmean<<<64, 256, 0, stream>>>(F[2], BRI0_, NU_ - BRI0_, &scal[16]);
  k_gterms_soc<<<1, 64, 0, stream>>>(&scal[16], &scal[3], &scal[4]);
  k_gmeans_sig<<<64, 256, 0, stream>>>(F[3], i_bri, i_bri_pos, &scal[80], &scal[144]);
  k_dgi_pair<<<NBRI_ * 64 / 256, 256, 0, stream>>>(F[0], F[1], i_bri, &scal[5], &scal[6]);
  k_gterms_infor<<<1, 64, 0, stream>>>(&scal[80], &scal[144], &scal[7], &scal[8]);
  k_finalize<<<1, 1, 0, stream>>>(scal, (float*)d_out);
}
